// GCNRegressor_27986006901220
// MI455X (gfx1250) — matrix operand order repaired and checked by reading, not compiled, not run on hardware
//
#include <hip/hip_runtime.h>
#include <hip/hip_bf16.h>

typedef __attribute__((ext_vector_type(16))) _Float16 v16h;
typedef __attribute__((ext_vector_type(8)))  _Float16 v8h;
typedef __attribute__((ext_vector_type(4)))  _Float16 v4h;
typedef __attribute__((ext_vector_type(8)))  float    v8f;

#define HID       128
#define NGRAPH    512

// ---------------------------------------------------------------- utilities
__global__ void k_zero_f(float* p, int n) {
    int i = blockIdx.x * blockDim.x + threadIdx.x;
    if (i < n) p[i] = 0.0f;
}

__global__ void k_fill_i(int* p, int v, int n) {
    int i = blockIdx.x * blockDim.x + threadIdx.x;
    if (i < n) p[i] = v;
}

// counts[col[e]] += 1   (counts pre-filled with 1 for the self-loop)
__global__ void k_count(const int* __restrict__ ei, int* __restrict__ counts, int E) {
    int e = blockIdx.x * blockDim.x + threadIdx.x;
    if (e < E) atomicAdd(&counts[ei[E + e]], 1);
}

// dis[i] = rsqrt(deg)  (deg >= 1 always, self-loop included)
__global__ void k_dis(const int* __restrict__ counts, float* __restrict__ dis, int n) {
    int i = blockIdx.x * blockDim.x + threadIdx.x;
    if (i < n) {
        float d = (float)counts[i];
        dis[i] = rsqrtf(fmaxf(d, 1.0f));
    }
}

// single-block exclusive scan of counts -> offs (+ copy into cursor)
__global__ void k_scan(const int* __restrict__ counts, int* __restrict__ offs,
                       int* __restrict__ cursor, int n) {
    __shared__ int sums[1024];
    int tid = threadIdx.x;
    int chunk = (n + 1023) >> 10;
    int beg = tid * chunk;
    int end = min(beg + chunk, n);
    int s = 0;
    for (int i = beg; i < end; ++i) s += counts[i];
    sums[tid] = s;
    __syncthreads();
    for (int off = 1; off < 1024; off <<= 1) {
        int v = 0;
        if (tid >= off) v = sums[tid - off];
        __syncthreads();
        if (tid >= off) sums[tid] += v;
        __syncthreads();
    }
    int prefix = (tid == 0) ? 0 : sums[tid - 1];
    for (int i = beg; i < end; ++i) {
        offs[i]   = prefix;
        cursor[i] = prefix;
        prefix   += counts[i];
    }
}

// fill interleaved CSR (row, norm-bits): edges [0,E) plus self-loops [E, E+N)
__global__ void k_fill_csr(const int* __restrict__ ei, const float* __restrict__ dis,
                           int* __restrict__ cursor, int2* __restrict__ csr, int E, int n) {
    int idx = blockIdx.x * blockDim.x + threadIdx.x;
    if (idx < E) {
        int r = ei[idx];
        int c = ei[E + idx];
        int pos = atomicAdd(&cursor[c], 1);
        csr[pos] = make_int2(r, __float_as_int(dis[r] * dis[c]));
    } else if (idx < E + n) {
        int node = idx - E;
        int pos = atomicAdd(&cursor[node], 1);
        csr[pos] = make_int2(node, __float_as_int(dis[node] * dis[node]));
    }
}

// plain f32 -> f16 conversion (node features x)
__global__ void k_cvt_h(const float* __restrict__ src, _Float16* __restrict__ dst, int n) {
    int i = blockIdx.x * blockDim.x + threadIdx.x;
    if (i < n) dst[i] = (_Float16)src[i];
}

// f32 W[K=128][N=128] -> f16 in WMMA B-fragment order:
// Wt[(((kb*2+half)*128)+n)*16 + i] = W[(kb*32 + half*16 + i)*128 + n]
// so each lane's 16 halves are contiguous (32B) -> 2x global_load_b128.
__global__ void k_cvt_wt(const float* __restrict__ W, _Float16* __restrict__ Wt) {
    int idx = blockIdx.x * blockDim.x + threadIdx.x;   // 16384 total
    int i  = idx & 15;
    int n  = (idx >> 4) & 127;
    int hb = (idx >> 11) & 1;
    int kb = idx >> 12;
    int K  = kb * 32 + hb * 16 + i;
    Wt[idx] = (_Float16)W[K * HID + n];
}

// -------------------------------------------------- WMMA GEMM: Hout = Hin @ W (+bias)(relu)
// Hin: [M,128] f16, Wt: swizzled f16 weights, Hout: [M,128] f16.
// Block = 256 threads (8 waves); block owns a 16-row strip, wave w owns cols [16w,16w+16).
// BIAS/RELU are compile-time so the epilogue is branch-free.
template <bool BIAS, bool RELU>
__global__ void __launch_bounds__(256)
k_gemm_wmma(const _Float16* __restrict__ Hin, const _Float16* __restrict__ Wt,
            const float* __restrict__ bias, _Float16* __restrict__ Hout) {
    __shared__ _Float16 lds[16 * HID];
    int m0  = blockIdx.x * 16;
    int tid = threadIdx.x;

    // stage the 16x128 f16 strip: 8 halves (16B) per thread
    *(v8h*)(lds + tid * 8) = *(const v8h*)(Hin + (size_t)m0 * HID + tid * 8);
    __syncthreads();

    int wave = tid >> 5, lane = tid & 31;
    int half = lane >> 4, lanelo = lane & 15;
    int n0 = wave * 16;

    v8f acc = {};
    #pragma unroll
    for (int kb = 0; kb < 4; ++kb) {
        int kk = kb * 32;
        v16h a;
        #pragma unroll
        for (int v = 0; v < 8; ++v) {
            int koff = kk + 16 * half + 2 * v;
            a[2 * v]     = lds[lanelo * HID + koff];
            a[2 * v + 1] = lds[lanelo * HID + koff + 1];
        }
        v16h b = *(const v16h*)(Wt + ((size_t)(kb * 2 + half) * HID + n0 + lanelo) * 16);
        acc = __builtin_amdgcn_wmma_f32_16x16x32_f16(
            false, a, false, b, (short)0, acc, false, false);
    }

    int col = n0 + lanelo;
    float bv = 0.0f;
    if (BIAS) bv = bias[col];
    #pragma unroll
    for (int r = 0; r < 8; ++r) {
        int row = m0 + r + 8 * half;
        float v = acc[r];
        if (BIAS) v += bv;
        if (RELU) v = fmaxf(v, 0.0f);
        Hout[(size_t)row * HID + col] = (_Float16)v;
    }
}

// ------------------- CSR gather aggregation: out[n] = relu(sum norm*T[row] + b)  (f16 in/out)
// one 32-thread block per node -> csr/offs/counts addresses are wave-uniform (scalar loads).
__global__ void __launch_bounds__(32)
k_aggregate(const int2* __restrict__ csr, const int* __restrict__ offs,
            const int* __restrict__ counts, const _Float16* __restrict__ T,
            const float* __restrict__ bias, _Float16* __restrict__ out) {
    int node = blockIdx.x;
    int lane = threadIdx.x;
    int beg = offs[node];
    int cnt = counts[node];
    float ax = 0.f, ay = 0.f, az = 0.f, aw = 0.f;
    #pragma unroll 4
    for (int i = 0; i < cnt; ++i) {
        int2 e = csr[beg + i];               // uniform -> scalar load
        if (i + 8 < cnt) {                   // hide L2 latency on the random gather
            int rp = csr[beg + i + 8].x;
            __builtin_prefetch((const void*)(T + (size_t)rp * HID + lane * 4), 0, 1);
        }
        float w = __int_as_float(e.y);
        v4h v = *(const v4h*)(T + (size_t)e.x * HID + lane * 4);
        ax += w * (float)v[0];
        ay += w * (float)v[1];
        az += w * (float)v[2];
        aw += w * (float)v[3];
    }
    const float4 b4 = *(const float4*)(bias + lane * 4);
    ax = fmaxf(ax + b4.x, 0.f);
    ay = fmaxf(ay + b4.y, 0.f);
    az = fmaxf(az + b4.z, 0.f);
    aw = fmaxf(aw + b4.w, 0.f);
    v4h o;
    o[0] = (_Float16)ax; o[1] = (_Float16)ay; o[2] = (_Float16)az; o[3] = (_Float16)aw;
    *(v4h*)(out + (size_t)node * HID + lane * 4) = o;
}

// ------------------------------------------------------- pooling (segment sum + count)
__global__ void __launch_bounds__(32)
k_pool(const _Float16* __restrict__ H, const int* __restrict__ batch,
       float* __restrict__ pooled, int* __restrict__ cnt) {
    int node = blockIdx.x;
    int lane = threadIdx.x;
    int g = batch[node];
    v4h v = *(const v4h*)(H + (size_t)node * HID + lane * 4);
    float* dst = pooled + (size_t)g * HID + lane * 4;
    unsafeAtomicAdd(dst + 0, (float)v[0]);
    unsafeAtomicAdd(dst + 1, (float)v[1]);
    unsafeAtomicAdd(dst + 2, (float)v[2]);
    unsafeAtomicAdd(dst + 3, (float)v[3]);
    if (lane == 0) atomicAdd(&cnt[g], 1);
}

__global__ void k_mean(const float* __restrict__ pooled, const int* __restrict__ cnt,
                       _Float16* __restrict__ gmean, int total) {
    int i = blockIdx.x * blockDim.x + threadIdx.x;
    if (i < total) {
        int g = i >> 7;
        float c = fmaxf((float)cnt[g], 1.0f);
        gmean[i] = (_Float16)(pooled[i] / c);
    }
}

// ------------------------------------------------------- final 128 -> 1 projection
__global__ void k_final(const _Float16* __restrict__ gh, const float* __restrict__ Wm1,
                        const float* __restrict__ bm1, float* __restrict__ out, int G) {
    int row = blockIdx.x * (blockDim.x >> 5) + (threadIdx.x >> 5);
    if (row >= G) return;
    int lane = threadIdx.x & 31;
    v4h v = *(const v4h*)(gh + (size_t)row * HID + lane * 4);
    const float4 w = *(const float4*)(Wm1 + lane * 4);
    float acc = (float)v[0] * w.x + (float)v[1] * w.y + (float)v[2] * w.z + (float)v[3] * w.w;
    #pragma unroll
    for (int off = 16; off > 0; off >>= 1)
        acc += __shfl_down(acc, off, 32);
    if (lane == 0) out[row] = acc + bm1[0];
}

// ---------------------------------------------------------------- launch
extern "C" void kernel_launch(void* const* d_in, const int* in_sizes, int n_in,
                              void* d_out, int out_size, void* d_ws, size_t ws_size,
                              hipStream_t stream) {
    const float* x    = (const float*)d_in[0];
    const int*   ei   = (const int*)d_in[1];
    // d_in[2] = edge_attr (unused, matching reference)
    const int*   batch= (const int*)d_in[3];
    const float* W0   = (const float*)d_in[4];
    const float* b0   = (const float*)d_in[5];
    const float* W1   = (const float*)d_in[6];
    const float* b1   = (const float*)d_in[7];
    const float* W2   = (const float*)d_in[8];
    const float* b2   = (const float*)d_in[9];
    const float* Wm0  = (const float*)d_in[10];
    const float* bm0  = (const float*)d_in[11];
    const float* Wm1  = (const float*)d_in[12];
    const float* bm1  = (const float*)d_in[13];
    float* out = (float*)d_out;

    const int N = in_sizes[0] / HID;      // 100000
    const int E = in_sizes[1] / 2;        // 3200000
    const int G = NGRAPH;                 // 512
    const int EN = E + N;

    // workspace layout (256B aligned segments)
    auto au = [](size_t v) { return (v + 255) & ~(size_t)255; };
    char* p = (char*)d_ws;
    float* dis      = (float*)p; p += au((size_t)N * 4);
    int*   counts   = (int*)p;   p += au((size_t)N * 4);
    int*   offs     = (int*)p;   p += au((size_t)N * 4);
    int*   cursor   = (int*)p;   p += au((size_t)N * 4);
    int2*  csr      = (int2*)p;  p += au((size_t)EN * 8);
    _Float16* xh    = (_Float16*)p; p += au((size_t)N * HID * 2);
    _Float16* Hb0   = (_Float16*)p; p += au((size_t)N * HID * 2);
    _Float16* Hb1   = (_Float16*)p; p += au((size_t)N * HID * 2);
    _Float16* Wt0   = (_Float16*)p; p += au((size_t)HID * HID * 2);
    _Float16* Wt1   = (_Float16*)p; p += au((size_t)HID * HID * 2);
    _Float16* Wt2   = (_Float16*)p; p += au((size_t)HID * HID * 2);
    _Float16* Wtm0  = (_Float16*)p; p += au((size_t)HID * HID * 2);
    float* pooled   = (float*)p; p += au((size_t)G * HID * 4);
    int*   cnt      = (int*)p;   p += au((size_t)G * 4);
    _Float16* gmean = (_Float16*)p; p += au((size_t)G * HID * 2);
    _Float16* gh    = (_Float16*)p; p += au((size_t)G * HID * 2);
    (void)ws_size; (void)n_in; (void)out_size;

    const int T256 = 256;
    // ---- normalization + CSR build (once per call, amortized over 3 layers)
    k_fill_i<<<(N + T256 - 1) / T256, T256, 0, stream>>>(counts, 1, N);
    k_count<<<(E + T256 - 1) / T256, T256, 0, stream>>>(ei, counts, E);
    k_dis<<<(N + T256 - 1) / T256, T256, 0, stream>>>(counts, dis, N);
    k_scan<<<1, 1024, 0, stream>>>(counts, offs, cursor, N);
    k_fill_csr<<<(EN + T256 - 1) / T256, T256, 0, stream>>>(ei, dis, cursor, csr, E, N);

    // ---- one-time conversions: x -> f16, weights -> swizzled f16
    k_cvt_h<<<((N * HID) + T256 - 1) / T256, T256, 0, stream>>>(x, xh, N * HID);
    k_cvt_wt<<<(HID * HID) / T256, T256, 0, stream>>>(W0,  Wt0);
    k_cvt_wt<<<(HID * HID) / T256, T256, 0, stream>>>(W1,  Wt1);
    k_cvt_wt<<<(HID * HID) / T256, T256, 0, stream>>>(W2,  Wt2);
    k_cvt_wt<<<(HID * HID) / T256, T256, 0, stream>>>(Wm0, Wtm0);

    const int gemmBlocks = N / 16;   // 6250 (N multiple of 16)

    // ---- layer 0: T = xh @ W0 ; H = relu(agg(T) + b0)
    k_gemm_wmma<false, false><<<gemmBlocks, T256, 0, stream>>>(xh, Wt0, nullptr, Hb1);
    k_aggregate<<<N, 32, 0, stream>>>(csr, offs, counts, Hb1, b0, Hb0);
    // ---- layer 1
    k_gemm_wmma<false, false><<<gemmBlocks, T256, 0, stream>>>(Hb0, Wt1, nullptr, Hb1);
    k_aggregate<<<N, 32, 0, stream>>>(csr, offs, counts, Hb1, b1, Hb0);
    // ---- layer 2
    k_gemm_wmma<false, false><<<gemmBlocks, T256, 0, stream>>>(Hb0, Wt2, nullptr, Hb1);
    k_aggregate<<<N, 32, 0, stream>>>(csr, offs, counts, Hb1, b2, Hb0);

    // ---- mean pool over graphs
    k_zero_f<<<(G * HID + T256 - 1) / T256, T256, 0, stream>>>(pooled, G * HID);
    k_fill_i<<<(G + T256 - 1) / T256, T256, 0, stream>>>(cnt, 0, G);
    k_pool<<<N, 32, 0, stream>>>(Hb0, batch, pooled, cnt);
    k_mean<<<(G * HID + T256 - 1) / T256, T256, 0, stream>>>(pooled, cnt, gmean, G * HID);

    // ---- MLP head: gh = relu(gmean @ Wm0 + bm0) ; out = gh @ Wm1 + bm1
    k_gemm_wmma<true, true><<<G / 16, T256, 0, stream>>>(gmean, Wtm0, bm0, gh);
    k_final<<<(G + 7) / 8, T256, 0, stream>>>(gh, Wm1, bm1, out, G);
}
